// MrbConv2d_17910013624549
// MI455X (gfx1250) — matrix operand order repaired and checked by reading, not compiled, not run on hardware
//
#include <hip/hip_runtime.h>
#include <stdint.h>
#include <stddef.h>

// ---------------------------------------------------------------------------
// Binary-weight multi-dilation conv bank for MI455X (gfx1250), wave32 WMMA.
// GEMM view per conv: M=64 (co), K=256c*5taps, N=one h row (128 w).
// x is split ONCE per staged LDS element into packed (bf16 hi << 16)|bf16 lo;
// +/-1 weights make both products exact, f32 WMMA accumulation -> fp32-grade.
// ---------------------------------------------------------------------------

typedef __attribute__((ext_vector_type(16))) __bf16 v16bf;
typedef __attribute__((ext_vector_type(8)))  float  v8f;
typedef __attribute__((ext_vector_type(8)))  int    i32x8;
typedef __attribute__((ext_vector_type(4)))  int    i32x4;

#if __has_builtin(__builtin_amdgcn_global_load_async_to_lds_b128)
#define ASYNC_LDS 1
// exact parameter types per clang's prototype: AS(1)/AS(3) int4 pointers
typedef __attribute__((address_space(1))) i32x4* as1_i32x4_p;
typedef __attribute__((address_space(3))) i32x4* as3_i32x4_p;
#else
#define ASYNC_LDS 0
#endif

#define CC      16    // channels staged per K-chunk
#define RSTRIDE 132   // 128 words + 4 pad -> 528B rows (16B aligned, de-conflicted)

template <int VERT, int DIL>
__global__ __launch_bounds__(256) void bconv4_wmma(
    const float* __restrict__ x,     // (16, 256, 128, 128) f32
    const float* __restrict__ wgt,   // (64, 256, 5) f32
    const float* __restrict__ bias,  // (64,) f32
    float* __restrict__ out,         // (16, 256, 128, 128) f32
    int cobase)
{
  constexpr int NROW = VERT ? 5 : 1;       // rows staged per channel
  constexpr int NELT = CC * NROW * 128;    // staged elements per chunk
  constexpr int NVEC = NELT / 4;           // float4 groups per chunk

  const int h    = blockIdx.x;             // output row 0..127
  const int n    = blockIdx.y;             // image 0..15
  const int tid  = threadIdx.x;
  const int lane = tid & 31;
  const int wv   = tid >> 5;               // wave 0..7
  const int cot  = wv & 3;                 // co tile (16 co each)
  const int whalf= wv >> 2;                // which 64-wide half of the row
  const int lHi  = lane >> 4;              // K-half select per ISA layout
  const int l15  = lane & 15;

  __shared__ unsigned int xs[CC * NROW * RSTRIDE];  // 8.4 KB (horiz) / 42.2 KB (vert)

  v8f acc[4] = {};                         // 4 C tiles of 16x16 f32 per wave
  const int coA = (cot << 4) + l15;        // A-matrix row: M = lane & 15

  for (int cc = 0; cc < 256; cc += CC) {
    __syncthreads();                       // previous chunk's LDS reads done

    // ---- stage CC channels x NROW rows x 128 f32 into LDS (async DMA) ----
#pragma unroll
    for (int k = 0; k < NVEC / 256; ++k) {
      const int i    = tid + k * 256;
      const int cl   = i / (NROW * 32);    // compile-time divisor
      const int rem  = i - cl * (NROW * 32);
      const int r    = rem >> 5;
      const int w4   = (rem & 31) << 2;
      const int ldso = (cl * NROW + r) * RSTRIDE + w4;
      const int hs   = h + (VERT ? (r - 2) * DIL : 0);
      if (!VERT || (hs >= 0 && hs < 128)) {
        const float* gp =
            x + ((size_t)((n * 256 + cc + cl) * 128 + hs) << 7) + w4;
#if ASYNC_LDS
        __builtin_amdgcn_global_load_async_to_lds_b128(
            (as1_i32x4_p)gp, (as3_i32x4_p)&xs[ldso], 0, 0);
#else
        *(i32x4*)&xs[ldso] = *(const i32x4*)gp;
#endif
      } else {
        const i32x4 z = {};                // 'SAME' zero padding rows
        *(i32x4*)&xs[ldso] = z;
      }
    }
#if ASYNC_LDS
#if __has_builtin(__builtin_amdgcn_s_wait_asynccnt)
    __builtin_amdgcn_s_wait_asynccnt(0);
#else
    asm volatile("s_wait_asynccnt 0" ::: "memory");
#endif
#endif
    __syncthreads();

    // ---- in-place split: f32 -> (bf16 hi << 16) | bf16 lo, 4-wide b128 ----
#pragma unroll
    for (int k = 0; k < NVEC / 256; ++k) {
      const int i    = tid + k * 256;      // float4-group index
      const int row  = i >> 5;             // 32 groups per 128-word row
      const int col4 = (i & 31) << 2;
      const int o    = row * RSTRIDE + col4;
      i32x4 w = *(i32x4*)&xs[o];
#pragma unroll
      for (int e = 0; e < 4; ++e) {
        const float v = __builtin_bit_cast(float, w[e]);
        const unsigned int hu = __builtin_bit_cast(unsigned short, (__bf16)v);
        const float hf = __builtin_bit_cast(float, hu << 16);
        const unsigned int lu =
            __builtin_bit_cast(unsigned short, (__bf16)(v - hf));
        w[e] = (int)((hu << 16) | lu);
      }
      *(i32x4*)&xs[o] = w;
    }
    __syncthreads();

    // ---- K = 16 channels x 2 taps per WMMA; 3 tap-pairs cover 5 taps ----
#pragma unroll
    for (int tp = 0; tp < 3; ++tp) {
      const int t0 = tp * 2;

      // A fragment: binarized weight signs as packed bf16 words
      // (A layout matches B: lane<16 holds channels 0..7, lane>=16 channels 8..15).
      i32x8 aw;
      const int aTbl[8] = {0, 1, 2, 3, 8, 9, 10, 11};
#pragma unroll
      for (int v = 0; v < 8; ++v) {
        const int cl = v + (lHi ? 8 : 0);
        const float* wp =
            wgt + ((size_t)coA * 256 + (size_t)(cc + cl)) * 5 + t0;
        unsigned int word = (wp[0] < 0.f) ? 0xBF80u : 0x3F80u;  // bf16 +/-1
        if (t0 + 1 <= 4)
          word |= ((wp[1] < 0.f) ? 0xBF80u : 0x3F80u) << 16;    // else tap6 = +0
        aw[v] = (int)word;
      }
      const v16bf A = __builtin_bit_cast(v16bf, aw);

#pragma unroll
      for (int wt = 0; wt < 4; ++wt) {
        const int wpos = (((whalf << 2) + wt) << 4) + l15;  // N = lane & 15

        // B fragments: 2 packed LDS words -> hi-pair / lo-pair via v_perm_b32.
        i32x8 hw, lw;
#pragma unroll
        for (int v = 0; v < 8; ++v) {
          const int cl = v + (lHi ? 8 : 0);  // this code's k = 16h + s order (same as A): lane<16 ch 0..7, else ch 8..15
          unsigned int p0 = 0u, p1 = 0u;
          if (VERT) {
            p0 = xs[(cl * NROW + t0) * RSTRIDE + wpos];
            if (t0 + 1 <= 4)
              p1 = xs[(cl * NROW + t0 + 1) * RSTRIDE + wpos];
          } else {
            const int ws0 = wpos + (t0 - 2) * DIL;
            if (ws0 >= 0 && ws0 < 128) p0 = xs[cl * RSTRIDE + ws0];
            if (t0 + 1 <= 4) {
              const int ws1 = wpos + (t0 - 1) * DIL;
              if (ws1 >= 0 && ws1 < 128) p1 = xs[cl * RSTRIDE + ws1];
            }
          }
          hw[v] = (int)__builtin_amdgcn_perm(p1, p0, 0x07060302u);  // hi halves
          lw[v] = (int)__builtin_amdgcn_perm(p1, p0, 0x05040100u);  // lo halves
        }
        const v16bf Bh = __builtin_bit_cast(v16bf, hw);
        const v16bf Bl = __builtin_bit_cast(v16bf, lw);

        acc[wt] = __builtin_amdgcn_wmma_f32_16x16x32_bf16(
            false, A, false, Bh, (short)0, acc[wt], false, false);
        acc[wt] = __builtin_amdgcn_wmma_f32_16x16x32_bf16(
            false, A, false, Bl, (short)0, acc[wt], false, false);
      }
    }
  }

  // ---- epilogue: bias + store (C layout: VGPR r -> M=r+8*laneHi, N=lane&15) ----
  const int co0 = (cot << 4) + (lHi ? 8 : 0);
  float bv[8];
#pragma unroll
  for (int r = 0; r < 8; ++r) bv[r] = bias[co0 + r];

#pragma unroll
  for (int wt = 0; wt < 4; ++wt) {
    const int wpos = (((whalf << 2) + wt) << 4) + l15;
#pragma unroll
    for (int r = 0; r < 8; ++r) {
      const int coAbs = cobase + co0 + r;
      out[((size_t)((n * 256 + coAbs) * 128 + h) << 7) + wpos] =
          acc[wt][r] + bv[r];
    }
  }
}

// ---------------------------------------------------------------------------
// Host side: 4 launches (one template specialization per conv branch).
// d_in order: x, w1, b1, w2, b2, w3, b3, w4, b4
// ---------------------------------------------------------------------------
extern "C" void kernel_launch(void* const* d_in, const int* in_sizes, int n_in,
                              void* d_out, int out_size, void* d_ws, size_t ws_size,
                              hipStream_t stream) {
  (void)in_sizes; (void)n_in; (void)out_size; (void)d_ws; (void)ws_size;

  const float* x = (const float*)d_in[0];
  float* out = (float*)d_out;

  dim3 grid(128, 16);   // (h, n)
  dim3 block(256);      // 8 wave32

  // conv0: (1,5) dil1 | conv1: (1,5) dil2 | conv2: (5,1) dil1 | conv3: (5,1) dil2
  bconv4_wmma<0, 1><<<grid, block, 0, stream>>>(
      x, (const float*)d_in[1], (const float*)d_in[2], out, 0);
  bconv4_wmma<0, 2><<<grid, block, 0, stream>>>(
      x, (const float*)d_in[3], (const float*)d_in[4], out, 64);
  bconv4_wmma<1, 1><<<grid, block, 0, stream>>>(
      x, (const float*)d_in[5], (const float*)d_in[6], out, 128);
  bconv4_wmma<1, 2><<<grid, block, 0, stream>>>(
      x, (const float*)d_in[7], (const float*)d_in[8], out, 192);
}
